// SwitchTransformersSparseMLP_2018634629713
// MI455X (gfx1250) — matrix operand order repaired and checked by reading, not compiled, not run on hardware
//
#include <hip/hip_runtime.h>
#include <math.h>

// ---------------------------------------------------------------------------
// Switch-Transformer sparse MLP (top-1 MoE) for gfx1250 / MI455X.
// bf16 WMMA grouped-GEMM path: v_wmma_f32_16x16x32_bf16, f32 accumulate.
// ---------------------------------------------------------------------------

#define DEV __device__ __forceinline__

typedef __bf16 bf16_t;
typedef __attribute__((ext_vector_type(8)))  bf16_t bf16x8;
typedef __attribute__((ext_vector_type(16))) bf16_t bf16x16;
typedef __attribute__((ext_vector_type(8)))  float  f32x8;

// Problem shape (fixed by the reference)
constexpr int Bb = 2, Ss = 2048, Dd = 1024, Ee = 8, Ff = 4096;
constexpr int T  = Bb * Ss;                 // 4096 tokens

// GEMM tiling
constexpr int BM = 128;                     // token rows per block
constexpr int BN = 128;                     // output cols per block
constexpr int BK = 32;                      // K per WMMA step
constexpr int PITCH = BK + 8;               // LDS row pitch in halves (80 B): 16B-aligned, conflict-breaking

// f32 -> bf16 round-to-nearest-even
DEV unsigned short f2bf_u(float f) {
  unsigned u = __float_as_uint(f);
  u += 0x7FFFu + ((u >> 16) & 1u);
  return (unsigned short)(u >> 16);
}

// two f32 -> packed 2x bf16 dword (lo in [15:0], hi in [31:16]) via v_perm_b32
DEV unsigned pack2bf(float flo, float fhi) {
  unsigned lo = __float_as_uint(flo);
  unsigned hi = __float_as_uint(fhi);
  lo += 0x7FFFu + ((lo >> 16) & 1u);
  hi += 0x7FFFu + ((hi >> 16) & 1u);
  // dst = { hi[31:16], lo[31:16] }
  return __builtin_amdgcn_perm(hi, lo, 0x07060302u);
}

DEV float f4c(const float4& v, int j) { return reinterpret_cast<const float*>(&v)[j]; }

// ---------------------------------------------------------------------------
// Kernel 1: router + dispatch + x->bf16.
// One wave32 per token. 8 logits via strided dot product, butterfly-reduced.
// ---------------------------------------------------------------------------
__global__ void router_dispatch(const float* __restrict__ x,
                                const float* __restrict__ Wr,
                                float* __restrict__ pmax,
                                int* __restrict__ counts,
                                int* __restrict__ tlist,
                                unsigned short* __restrict__ xbf) {
  const int wave = threadIdx.x >> 5;
  const int lane = threadIdx.x & 31;
  const int t = blockIdx.x * 8 + wave;
  if (t >= T) return;

  const float* xr = x + (size_t)t * Dd;
  float acc[Ee];
#pragma unroll
  for (int e = 0; e < Ee; ++e) acc[e] = 0.0f;

  for (int d = lane; d < Dd; d += 32) {
    float xv = xr[d];
    xbf[(size_t)t * Dd + d] = f2bf_u(xv);      // emit bf16 activations
    const float4* w = (const float4*)(Wr + (size_t)d * Ee);  // Wr row: 8 f32, 32B aligned
    float4 w0 = w[0], w1 = w[1];
    acc[0] += xv * w0.x; acc[1] += xv * w0.y; acc[2] += xv * w0.z; acc[3] += xv * w0.w;
    acc[4] += xv * w1.x; acc[5] += xv * w1.y; acc[6] += xv * w1.z; acc[7] += xv * w1.w;
  }

  // wave32 butterfly reduction for each of the 8 logits
#pragma unroll
  for (int e = 0; e < Ee; ++e) {
#pragma unroll
    for (int m = 16; m >= 1; m >>= 1)
      acc[e] += __shfl_xor(acc[e], m, 32);
  }

  if (lane == 0) {
    float mx = acc[0]; int arg = 0;
#pragma unroll
    for (int e = 1; e < Ee; ++e)
      if (acc[e] > mx) { mx = acc[e]; arg = e; }
    float s = 0.0f;
#pragma unroll
    for (int e = 0; e < Ee; ++e) s += expf(acc[e] - mx);
    pmax[t] = 1.0f / s;                         // max softmax prob = exp(0)/sum
    int pos = atomicAdd(&counts[arg], 1);
    tlist[arg * T + pos] = t;
  }
}

// ---------------------------------------------------------------------------
// Kernel 2/3: grouped GEMM per expert with bf16 WMMA.
//   GELU=true : h = gelu(x_bf16[gather] @ W1[e] + b1[e])        -> bf16 h
//   GELU=false: out = pmax * (h_bf16[gather] @ W2[e] + b2[e])   -> f32 out
// Block = 256 threads = 8 waves; block tile 128x128; wave tile 64x32 = 4x2
// WMMA fragments; software-pipelined global->reg->LDS staging.
// ---------------------------------------------------------------------------
template <bool GELU>
__global__ void __launch_bounds__(256)
moe_gemm(const unsigned short* __restrict__ A,   // [T, K] bf16 (gathered by token id)
         const float* __restrict__ Bw,           // [E, K, N] f32 weights
         const float* __restrict__ bias,         // [E, N]
         const int* __restrict__ counts,
         const int* __restrict__ tlist,
         const float* __restrict__ pmax,
         unsigned short* __restrict__ out_bf,    // GELU path
         float* __restrict__ out_f,              // combine path
         int K, int N) {
  const int e = blockIdx.z;
  const int mcount = counts[e];
  const int m0 = blockIdx.y * BM;
  if (m0 >= mcount) return;                      // empty tile for this expert
  const int n0 = blockIdx.x * BN;

  const int tid  = threadIdx.x;
  const int lane = tid & 31;
  const int wave = tid >> 5;

  __shared__ unsigned short sA[BM * PITCH];      // [m][k] row-major, bf16
  __shared__ unsigned short sB[BN * PITCH];      // [n][k] (transposed!), bf16
  __shared__ int stok[BM];

  if (tid < BM) {
    int row = m0 + tid;
    stok[tid] = (row < mcount) ? tlist[e * T + row] : tlist[e * T];  // clamp to a valid token
  }
  __syncthreads();

  const float* Bexp = Bw + (size_t)e * K * N;

  const int wm0 = (wave & 1) * 64;               // wave grid: 2 (M) x 4 (N)
  const int wn0 = (wave >> 1) * 32;

  f32x8 acc[4][2];
#pragma unroll
  for (int im = 0; im < 4; ++im)
#pragma unroll
    for (int in = 0; in < 2; ++in)
#pragma unroll
      for (int v = 0; v < 8; ++v) acc[im][in][v] = 0.0f;

  // A staging: 128 rows x 32 halves; 2 threads/row, 16 halves (32B) each
  const int rowA = tid >> 1;
  const int kcA  = (tid & 1) * 16;
  const unsigned short* Arow = A + (size_t)stok[rowA] * K;

  // B staging: k-pair kp = tid>>5 (k = 2*kp and +16), 4 columns nc..nc+3
  const int kp = tid >> 5;                       // 0..7
  const int nc = (tid & 31) * 4;                 // 0..124

  uint4  ra0, ra1;
  float4 rb[4];

  auto load_global = [&](int k0) {
    const unsigned short* ap = Arow + k0 + kcA;
    ra0 = *(const uint4*)(ap);
    ra1 = *(const uint4*)(ap + 8);
    const float* bp = Bexp + (size_t)(k0 + 2 * kp) * N + n0 + nc;
    rb[0] = *(const float4*)(bp);
    rb[1] = *(const float4*)(bp + N);
    rb[2] = *(const float4*)(bp + (size_t)16 * N);
    rb[3] = *(const float4*)(bp + (size_t)17 * N);
  };

  load_global(0);

  for (int k0 = 0; k0 < K; k0 += BK) {
    // ---- registers -> LDS ----
    *(uint4*)(&sA[rowA * PITCH + kcA])     = ra0;
    *(uint4*)(&sA[rowA * PITCH + kcA + 8]) = ra1;
    unsigned* sB32 = (unsigned*)sB;
#pragma unroll
    for (int j = 0; j < 4; ++j) {
      sB32[(nc + j) * (PITCH / 2) + kp]     = pack2bf(f4c(rb[0], j), f4c(rb[1], j));
      sB32[(nc + j) * (PITCH / 2) + 8 + kp] = pack2bf(f4c(rb[2], j), f4c(rb[3], j));
    }
    __syncthreads();

    // ---- prefetch next K-slab into registers (overlaps with WMMA) ----
    if (k0 + BK < K) load_global(k0 + BK);

    // ---- WMMA fragments straight out of LDS ----
    const int mrow = lane & 15;
    const int kh = (lane >> 4) * 8;              // A: lanes 16-31 take K+8 / K+24 chunks
    const int kb = (lane >> 4) * 16;             // unused: B loads also use kh (same K chunks as A)

    bf16x16 afrag[4], bfrag[2];
#pragma unroll
    for (int im = 0; im < 4; ++im) {
      const unsigned short* ar = sA + (size_t)(wm0 + im * 16 + mrow) * PITCH;
      bf16x8 lo = *(const bf16x8*)(ar + kh);          // K = kh .. kh+7
      bf16x8 hi = *(const bf16x8*)(ar + 16 + kh);     // K = 16+kh .. 16+kh+7
      afrag[im] = __builtin_shufflevector(lo, hi, 0,1,2,3,4,5,6,7,8,9,10,11,12,13,14,15);
    }
#pragma unroll
    for (int in = 0; in < 2; ++in) {
      const unsigned short* br = sB + (size_t)(wn0 + in * 16 + mrow) * PITCH;
      bf16x8 lo = *(const bf16x8*)(br + kh);          // K = kh .. kh+7
      bf16x8 hi = *(const bf16x8*)(br + 16 + kh);     // K = 16+kh .. 16+kh+7
      bfrag[in] = __builtin_shufflevector(lo, hi, 0,1,2,3,4,5,6,7,8,9,10,11,12,13,14,15);
    }

#pragma unroll
    for (int im = 0; im < 4; ++im)
#pragma unroll
      for (int in = 0; in < 2; ++in)
        acc[im][in] = __builtin_amdgcn_wmma_f32_16x16x32_bf16(
            /*neg_a=*/false, afrag[im], /*neg_b=*/false, bfrag[in],
            /*c_mod=*/(short)0, acc[im][in], /*reuse_a=*/false, /*reuse_b=*/false);

    __syncthreads();
  }

  // ---- epilogue: C layout = lanes 0-15 -> N=lane, M=v ; lanes 16-31 -> M=8+v ----
#pragma unroll
  for (int im = 0; im < 4; ++im) {
#pragma unroll
    for (int in = 0; in < 2; ++in) {
      const int ncol = n0 + wn0 + in * 16 + (lane & 15);
      const float bv = bias[(size_t)e * N + ncol];
#pragma unroll
      for (int v = 0; v < 8; ++v) {
        const int mloc = wm0 + im * 16 + (lane >> 4) * 8 + v;
        if (m0 + mloc < mcount) {
          const int tok = stok[mloc];
          float val = acc[im][in][v] + bv;
          if (GELU) {
            // exact GELU: 0.5*x*(1+erf(x/sqrt(2)))
            float g = 0.5f * val * (1.0f + erff(val * 0.70710678118654752f));
            out_bf[(size_t)tok * N + ncol] = f2bf_u(g);
          } else {
            out_f[(size_t)tok * N + ncol] = pmax[tok] * val;
          }
        }
      }
    }
  }
}

// ---------------------------------------------------------------------------
// Host launcher
// ---------------------------------------------------------------------------
extern "C" void kernel_launch(void* const* d_in, const int* in_sizes, int n_in,
                              void* d_out, int out_size, void* d_ws, size_t ws_size,
                              hipStream_t stream) {
  const float* x  = (const float*)d_in[0];   // [B,S,D]
  const float* Wr = (const float*)d_in[1];   // [D,E]
  const float* W1 = (const float*)d_in[2];   // [E,D,F]
  const float* b1 = (const float*)d_in[3];   // [E,F]
  const float* W2 = (const float*)d_in[4];   // [E,F,D]
  const float* b2 = (const float*)d_in[5];   // [E,D]
  float* out = (float*)d_out;                // [B,S,D]

  // workspace layout (all sections 256B aligned)
  char* ws = (char*)d_ws;
  constexpr size_t off_counts = 0;
  constexpr size_t off_tlist  = 256;
  constexpr size_t off_pmax   = off_tlist + (size_t)Ee * T * sizeof(int);
  constexpr size_t off_xbf    = (off_pmax + T * sizeof(float) + 255) & ~255ULL;
  constexpr size_t off_hbf    = (off_xbf + (size_t)T * Dd * 2 + 255) & ~255ULL;

  int*            counts = (int*)(ws + off_counts);
  int*            tlist  = (int*)(ws + off_tlist);
  float*          pmax   = (float*)(ws + off_pmax);
  unsigned short* xbf    = (unsigned short*)(ws + off_xbf);
  unsigned short* hbf    = (unsigned short*)(ws + off_hbf);

  hipMemsetAsync(counts, 0, Ee * sizeof(int), stream);

  // 1) router + dispatch + x->bf16   (T/8 blocks of 8 waves)
  router_dispatch<<<T / 8, 256, 0, stream>>>(x, Wr, pmax, counts, tlist, xbf);

  // 2) h = gelu(x @ W1[e] + b1[e])   — grouped GEMM, K=D=1024, N=F=4096
  dim3 g1(Ff / BN, (T + BM - 1) / BM, Ee);
  moe_gemm<true><<<g1, 256, 0, stream>>>(xbf, W1, b1, counts, tlist, pmax,
                                         hbf, nullptr, Dd, Ff);

  // 3) out = pmax * (h @ W2[e] + b2[e]) — grouped GEMM, K=F=4096, N=D=1024
  dim3 g2(Dd / BN, (T + BM - 1) / BM, Ee);
  moe_gemm<false><<<g2, 256, 0, stream>>>(hbf, W2, b2, counts, tlist, pmax,
                                          nullptr, out, Ff, Dd);
}
